// MixerBlock_63720134804187
// MI455X (gfx1250) — matrix operand order repaired and checked by reading, not compiled, not run on hardware
//
#include <hip/hip_runtime.h>
#include <hip/hip_bf16.h>
#include <stdint.h>

// ---------------------------------------------------------------------------
// Problem dimensions (fixed by the reference)
// ---------------------------------------------------------------------------
#define B_SZ   2
#define LSEQ   1024
#define D_OUT  1024      // D_OUTER
#define D_IN   2048      // D
#define NSTATE 16        // N
#define KCONV  4
#define PPN    (2*NSTATE + D_IN)   // 2080
#define MROWS  (B_SZ*LSEQ)         // 2048

typedef __attribute__((ext_vector_type(16))) _Float16 v16h;
typedef __attribute__((ext_vector_type(8)))  float    v8f;

struct alignas(16) h8 { _Float16 v[8]; };

// ---------------------------------------------------------------------------
// f32 -> f16 conversion (weights)
// ---------------------------------------------------------------------------
__global__ void cvt_f32_to_f16(const float* __restrict__ src,
                               _Float16* __restrict__ dst, int n) {
    int i = blockIdx.x * blockDim.x + threadIdx.x;
    if (i < n) dst[i] = (_Float16)src[i];
}

// ---------------------------------------------------------------------------
// LayerNorm over D_OUT, writes f16 (feeds WMMA GEMMs 1 & 3)
// ---------------------------------------------------------------------------
__global__ __launch_bounds__(256)
void ln_kernel(const float* __restrict__ X, const float* __restrict__ g,
               const float* __restrict__ beta, _Float16* __restrict__ Xn16) {
    __shared__ float s1[256], s2[256];
    const int row = blockIdx.x;              // 0..MROWS-1
    const float* xr = X + (size_t)row * D_OUT;
    float sum = 0.f, sq = 0.f;
    #pragma unroll
    for (int i = 0; i < D_OUT / 256; i++) {
        float v = xr[threadIdx.x + i * 256];
        sum += v; sq += v * v;
    }
    s1[threadIdx.x] = sum; s2[threadIdx.x] = sq;
    __syncthreads();
    for (int off = 128; off > 0; off >>= 1) {
        if (threadIdx.x < off) {
            s1[threadIdx.x] += s1[threadIdx.x + off];
            s2[threadIdx.x] += s2[threadIdx.x + off];
        }
        __syncthreads();
    }
    const float mu  = s1[0] * (1.f / D_OUT);
    const float var = s2[0] * (1.f / D_OUT) - mu * mu;
    const float rstd = rsqrtf(var + 1e-5f);
    _Float16* orow = Xn16 + (size_t)row * D_OUT;
    #pragma unroll
    for (int i = 0; i < D_OUT / 256; i++) {
        int c = threadIdx.x + i * 256;
        float v = (xr[c] - mu) * rstd * g[c] + beta[c];
        orow[c] = (_Float16)v;
    }
}

// ---------------------------------------------------------------------------
// Depthwise causal conv (K=4) + bias + SiLU.  Writes f32 (scan) and f16 (GEMM2)
// ---------------------------------------------------------------------------
__global__ __launch_bounds__(256)
void conv_silu_kernel(const float* __restrict__ Xm, const float* __restrict__ cw,
                      const float* __restrict__ cb,
                      float* __restrict__ out32, _Float16* __restrict__ out16) {
    int idx = blockIdx.x * blockDim.x + threadIdx.x;         // over B*L*D
    if (idx >= B_SZ * LSEQ * D_IN) return;
    const int d = idx & (D_IN - 1);
    const int l = (idx >> 11) & (LSEQ - 1);
    const int b = idx >> 21;
    float acc = cb[d];
    #pragma unroll
    for (int k = 0; k < KCONV; k++) {
        int lp = l - (KCONV - 1) + k;
        if (lp >= 0)
            acc += cw[d * KCONV + k] * Xm[(((size_t)b * LSEQ + lp) * D_IN) + d];
    }
    float s = acc / (1.f + __expf(-acc));                    // SiLU
    out32[idx] = s;
    out16[idx] = (_Float16)s;
}

// ---------------------------------------------------------------------------
// CDNA5 async global->LDS copy (ASYNCcnt-tracked), 16 bytes per lane.
// lds = workgroup-relative LDS byte address (generic ptr truncated to [31:0]).
// ---------------------------------------------------------------------------
__device__ inline void async_b128(uint32_t lds, const _Float16* g) {
    asm volatile("global_load_async_to_lds_b128 %0, %1, off"
                 :: "v"(lds), "v"((unsigned long long)(uintptr_t)g)
                 : "memory");
}
__device__ inline uint32_t lds_addr(const void* p) {
    return (uint32_t)(uintptr_t)p;
}

// ---------------------------------------------------------------------------
// WMMA fragment loaders (per CDNA5 ISA VGPR layouts, wave32)
// A: 16x32 f16, lane-half h: K chunks {h*8 .. h*8+7} and {16+h*8 ..}
// B: 32x16 f16 (N rows of W), lane n=lane&15, K chunks {h*8 .. h*8+7} and {16+h*8 ..}
// ---------------------------------------------------------------------------
__device__ inline v16h load_frag_a(const _Float16* base, int ld) {
    const int lane = threadIdx.x & 31;
    const int h = lane >> 4, r = lane & 15;
    const _Float16* p = base + r * ld + h * 8;
    h8 a0 = *(const h8*)p;
    h8 a1 = *(const h8*)(p + 16);
    v16h f;
    #pragma unroll
    for (int i = 0; i < 8; i++) { f[i] = a0.v[i]; f[8 + i] = a1.v[i]; }
    return f;
}
__device__ inline v16h load_frag_b(const _Float16* base, int ld) {
    const int lane = threadIdx.x & 31;
    const int h = lane >> 4, n = lane & 15;
    const _Float16* p = base + n * ld + h * 8;
    h8 a0 = *(const h8*)p;
    h8 a1 = *(const h8*)(p + 16);
    v16h f;
    #pragma unroll
    for (int i = 0; i < 8; i++) { f[i] = a0.v[i]; f[8 + i] = a1.v[i]; }
    return f;
}

// ---------------------------------------------------------------------------
// Tiled WMMA GEMM:  C[M,N] = A[M,K] * W[N,K]^T   (TN layout, f16 in / f32 out)
// Block tile 128x128, 8 waves (2x4), wave tile 64x32 -> 4x2 = 8 accumulators,
// 8 WMMAs per 32-K step.  Double-buffered LDS filled by ASYNC global->LDS
// copies; the K loop steady state is branch-free: prefetch next stage,
// one uniform s_wait_asynccnt 0x4, barrier, compute; final stage is peeled.
// Ragged-N W rows are CLAMPED (always in-bounds, uniform async count);
// duplicate/garbage rows only feed accumulator columns that are never stored.
// Epilogues: 0 none | 1 +bias | 2 silu | 3 +bias+residual
// ---------------------------------------------------------------------------
#define BM 128
#define BN 128
#define BK 32
#define LDT 40   // padded LDS row stride (halves); keeps 16B alignment

template <int EPI>
__global__ __launch_bounds__(256)
void gemm_wmma(const _Float16* __restrict__ A, const _Float16* __restrict__ W,
               float* __restrict__ C, int M, int N, int K,
               const float* __restrict__ bias, const float* __restrict__ res) {
    __shared__ _Float16 sA[2][BM * LDT];
    __shared__ _Float16 sB[2][BN * LDT];

    const int tid  = threadIdx.x;
    const int lane = tid & 31;
    const int wave = tid >> 5;
    const int waveM = wave & 1;          // 2 along M (64 rows each)
    const int waveN = wave >> 1;         // 4 along N (32 cols each)
    const int bM = blockIdx.x * BM;
    const int bN = blockIdx.y * BN;

    // tile-staging mapping: row = tid>>1, 32 bytes (two b128) at colh base
    const int srow = tid >> 1;           // 0..127
    const int scol = (tid & 1) * 16;     // halves: 0 / 16
    const int wrow = (bN + srow < N) ? (bN + srow) : (N - 1);  // clamped

    v8f acc[4][2];
    const v8f vzero = {0.f, 0.f, 0.f, 0.f, 0.f, 0.f, 0.f, 0.f};
    #pragma unroll
    for (int i = 0; i < 4; i++)
        #pragma unroll
        for (int j = 0; j < 2; j++) acc[i][j] = vzero;

    // ---- async stage loader: 4 b128 copies per thread (uniform count) ----
    auto stage_load = [&](int buf, int k0) {
        const _Float16* ga = A + (size_t)(bM + srow) * K + k0 + scol;
        uint32_t la = lds_addr(&sA[buf][srow * LDT + scol]);
        async_b128(la,      ga);
        async_b128(la + 16, ga + 8);          // +16 bytes / +8 halves
        const _Float16* gb = W + (size_t)wrow * K + k0 + scol;
        uint32_t lb = lds_addr(&sB[buf][srow * LDT + scol]);
        async_b128(lb,      gb);
        async_b128(lb + 16, gb + 8);
    };

    // ---- one 32-K compute stage: 6 ds fragment loads, 8 WMMAs ----
    auto compute = [&](int buf) {
        v16h af[4], bf[2];
        #pragma unroll
        for (int i = 0; i < 4; i++)
            af[i] = load_frag_a(&sA[buf][(waveM * 64 + i * 16) * LDT], LDT);
        #pragma unroll
        for (int j = 0; j < 2; j++)
            bf[j] = load_frag_b(&sB[buf][(waveN * 32 + j * 16) * LDT], LDT);
        #pragma unroll
        for (int i = 0; i < 4; i++)
            #pragma unroll
            for (int j = 0; j < 2; j++)
                acc[i][j] = __builtin_amdgcn_wmma_f32_16x16x32_f16(
                    false, af[i], false, bf[j], (short)0, acc[i][j],
                    false, false);
    };

    const int numK = K / BK;             // >= 2 for all four GEMMs
    stage_load(0, 0);

    for (int it = 0; it < numK - 1; ++it) {
        const int buf = it & 1;
        stage_load(buf ^ 1, (it + 1) * BK);            // prefetch next stage
        asm volatile("s_wait_asynccnt 0x4" ::: "memory");  // prev stage done
        __syncthreads();
        compute(buf);
        __syncthreads();                 // buf may be overwritten next iter
    }
    // peeled final stage
    asm volatile("s_wait_asynccnt 0x0" ::: "memory");
    __syncthreads();
    compute((numK - 1) & 1);

    // store: D layout -> col = lane&15, row = r + 8*(lane>>4)
    const int rbase = bM + waveM * 64;
    const int cbase = bN + waveN * 32;
    const int rOff = (lane >> 4) * 8;
    const int cOff = lane & 15;
    #pragma unroll
    for (int i = 0; i < 4; i++) {
        #pragma unroll
        for (int j = 0; j < 2; j++) {
            const int col = cbase + j * 16 + cOff;
            if (col >= N) continue;
            #pragma unroll
            for (int r = 0; r < 8; r++) {
                const int row = rbase + i * 16 + rOff + r;
                float v = acc[i][j][r];
                if (EPI == 1) v += bias[col];
                if (EPI == 2) v = v / (1.f + __expf(-v));
                if (EPI == 3) v += bias[col] + res[(size_t)row * N + col];
                C[(size_t)row * N + col] = v;
            }
        }
    }
}

// ---------------------------------------------------------------------------
// Selective scan: one lane per (b,d); H[16] and A[16] live in VGPRs.
// Reads pp (delta/B/C), X_act(f32), gate(f32, post-SiLU); writes Y_f16.
// ---------------------------------------------------------------------------
__global__ __launch_bounds__(256)
void scan_kernel(const float* __restrict__ pp, const float* __restrict__ xact,
                 const float* __restrict__ A_log, const float* __restrict__ gate,
                 _Float16* __restrict__ Y16) {
    const int tid = blockIdx.x * blockDim.x + threadIdx.x;   // 0..B*D-1
    if (tid >= B_SZ * D_IN) return;
    const int b = tid / D_IN;
    const int d = tid % D_IN;

    float A[NSTATE], H[NSTATE];
    #pragma unroll
    for (int n = 0; n < NSTATE; n++) {
        A[n] = -__expf(A_log[d * NSTATE + n]);
        H[n] = 0.f;
    }

    for (int t = 0; t < LSEQ; t++) {
        const size_t rowp = ((size_t)b * LSEQ + t);
        const float* prow = pp + rowp * PPN;
        if (t + 1 < LSEQ) {                     // hide latency of next step
            __builtin_prefetch(prow + PPN + d, 0, 1);
            __builtin_prefetch(prow + PPN + D_IN, 0, 1);
        }
        const float dl = prow[d];
        const float delta = (dl > 20.f) ? dl : log1pf(__expf(dl));  // softplus
        const float x = xact[rowp * D_IN + d];
        float y = 0.f;
        #pragma unroll
        for (int n = 0; n < NSTATE; n++) {
            const float dA = delta * A[n];
            const float ab = __expf(dA);
            const float zoh = (fabsf(dA) > 1e-6f) ? (ab - 1.f) / dA : 1.f;
            const float u = x * zoh * delta * prow[D_IN + n];
            H[n] = ab * H[n] + u;
            y += prow[D_IN + NSTATE + n] * H[n];
        }
        const float g = gate[rowp * D_IN + d];
        Y16[rowp * D_IN + d] = (_Float16)(y * g);
    }
}

// ---------------------------------------------------------------------------
// Host-side launcher
// ---------------------------------------------------------------------------
static inline size_t align256(size_t x) { return (x + 255) & ~(size_t)255; }

extern "C" void kernel_launch(void* const* d_in, const int* in_sizes, int n_in,
                              void* d_out, int out_size, void* d_ws, size_t ws_size,
                              hipStream_t stream) {
    const float* X      = (const float*)d_in[0];
    const float* ln_g   = (const float*)d_in[1];
    const float* ln_b   = (const float*)d_in[2];
    const float* W_up1  = (const float*)d_in[3];
    const float* conv_w = (const float*)d_in[4];
    const float* conv_b = (const float*)d_in[5];
    const float* W_ll   = (const float*)d_in[6];
    const float* b_ll   = (const float*)d_in[7];
    const float* A_log  = (const float*)d_in[8];
    const float* W_up2  = (const float*)d_in[9];
    const float* W_down = (const float*)d_in[10];
    const float* b_down = (const float*)d_in[11];
    float* out = (float*)d_out;

    // workspace layout
    char* ws = (char*)d_ws;
    size_t off = 0;
    auto alloc = [&](size_t bytes) { void* p = ws + off; off = align256(off + bytes); return p; };

    _Float16* Xn16   = (_Float16*)alloc((size_t)MROWS * D_OUT * 2);
    _Float16* Wup1h  = (_Float16*)alloc((size_t)D_IN * D_OUT * 2);
    _Float16* Wllh   = (_Float16*)alloc((size_t)PPN * D_IN * 2);
    _Float16* Wup2h  = (_Float16*)alloc((size_t)D_IN * D_OUT * 2);
    _Float16* Wdnh   = (_Float16*)alloc((size_t)D_OUT * D_IN * 2);
    float*    Xmain  = (float*)   alloc((size_t)MROWS * D_IN * 4);
    float*    Xact32 = (float*)   alloc((size_t)MROWS * D_IN * 4);
    _Float16* Xact16 = (_Float16*)alloc((size_t)MROWS * D_IN * 2);
    float*    ppbuf  = (float*)   alloc((size_t)MROWS * PPN * 4);
    float*    gatebf = (float*)   alloc((size_t)MROWS * D_IN * 4);
    _Float16* Y16    = (_Float16*)alloc((size_t)MROWS * D_IN * 2);
    (void)ws_size; (void)n_in; (void)in_sizes; (void)out_size;

    // 1) weight conversions f32 -> f16
    {
        int n1 = D_IN * D_OUT;        // W_up1 / W_up2 / W_down
        int n2 = PPN * D_IN;          // W_ll
        cvt_f32_to_f16<<<(n1 + 255) / 256, 256, 0, stream>>>(W_up1, Wup1h, n1);
        cvt_f32_to_f16<<<(n2 + 255) / 256, 256, 0, stream>>>(W_ll,  Wllh,  n2);
        cvt_f32_to_f16<<<(n1 + 255) / 256, 256, 0, stream>>>(W_up2, Wup2h, n1);
        cvt_f32_to_f16<<<(n1 + 255) / 256, 256, 0, stream>>>(W_down, Wdnh, n1);
    }

    // 2) LayerNorm
    ln_kernel<<<MROWS, 256, 0, stream>>>(X, ln_g, ln_b, Xn16);

    // 3) GEMM1: X_main = Xn @ W_up1^T         (2048 x 2048, K=1024)
    gemm_wmma<0><<<dim3(MROWS / BM, D_IN / BN), 256, 0, stream>>>(
        Xn16, Wup1h, Xmain, MROWS, D_IN, D_OUT, nullptr, nullptr);

    // 4) depthwise conv + bias + SiLU
    {
        int n = B_SZ * LSEQ * D_IN;
        conv_silu_kernel<<<(n + 255) / 256, 256, 0, stream>>>(
            Xmain, conv_w, conv_b, Xact32, Xact16);
    }

    // 5) GEMM2: pp = X_act @ W_ll^T + b_ll    (2048 x 2080, K=2048)
    gemm_wmma<1><<<dim3(MROWS / BM, (PPN + BN - 1) / BN), 256, 0, stream>>>(
        Xact16, Wllh, ppbuf, MROWS, PPN, D_IN, b_ll, nullptr);

    // 6) GEMM3: gate = silu(Xn @ W_up2^T)     (2048 x 2048, K=1024)
    gemm_wmma<2><<<dim3(MROWS / BM, D_IN / BN), 256, 0, stream>>>(
        Xn16, Wup2h, gatebf, MROWS, D_IN, D_OUT, nullptr, nullptr);

    // 7) selective scan -> Y16 = (scan output) * gate
    scan_kernel<<<(B_SZ * D_IN + 255) / 256, 256, 0, stream>>>(
        ppbuf, Xact32, A_log, gatebf, Y16);

    // 8) GEMM4: out = X + Y @ W_down^T + b_down   (2048 x 1024, K=2048)
    gemm_wmma<3><<<dim3(MROWS / BM, D_OUT / BN), 256, 0, stream>>>(
        Y16, Wdnh, out, MROWS, D_OUT, D_IN, b_down, X);
}
